// AdaptiveMPNN_71055938945660
// MI455X (gfx1250) — matrix operand order repaired and checked by reading, not compiled, not run on hardware
//
#include <hip/hip_runtime.h>
#include <hip/hip_bf16.h>

// ============================================================================
// AdaptiveMPNN for MI455X (gfx1250): f16 WMMA (fp32 accumulate) MLPs,
// one wave32 per 16-row tile, f16 LDS staging between layers, fused
// bias+ReLU epilogue in C-registers, vectorized LayerNorm epilogue,
// f32 atomics for segment-sum, fp32 masters for residuals.
//
// d_in layout assumption (top-level insertion order, nested params flattened
// via jax tree order = dict keys sorted alphabetically, lists in order):
//   0: x [50000,16] f32   1: edge_index [2,800000] int32   2: edge_features [800000,16] f32
//   3..50:  blocks[0..2], each: edge{Ws0(192x64),Ws1,Ws2,bs0,bs1,bs2,ln_b,ln_g}
//                               node{Ws0(128x64),Ws1,Ws2,bs0,bs1,bs2,ln_b,ln_g}
//   51..56: dec_edge {Ws0(64x64),Ws1(64x64),Ws2(64x8),bs0,bs1,bs2}
//   57..62: dec_node (same shapes)
//   63..70: enc_edge {Ws0(16x64),Ws1,Ws2,bs0,bs1,bs2,ln_b,ln_g}
//   71..78: enc_node (same)
// ============================================================================

typedef _Float16 f16;
typedef __attribute__((ext_vector_type(16))) _Float16 v16h;
typedef __attribute__((ext_vector_type(8)))  _Float16 v8h;
typedef __attribute__((ext_vector_type(8)))  float    v8f;

#define LN_EPS 1e-5f

// ---- fragment loaders (CDNA5 16-bit WMMA layouts, wave32) ------------------
// A 16x32 f16: lane L: row = L&15; j=0..7 -> k = hf*8+j ; j=8..15 -> k = 16+hf*8+(j-8)
__device__ __forceinline__ v16h load_a_frag(const f16* base, int row, int stride,
                                            int kbase, int hf) {
  const f16* p = base + row * stride + kbase + hf * 8;
  v8h lo = *(const v8h*)(p);
  v8h hi = *(const v8h*)(p + 16);
  v16h a;
#pragma unroll
  for (int i = 0; i < 8; ++i) { a[i] = lo[i]; a[i + 8] = hi[i]; }
  return a;
}

// B 32x16 f16: lane L: col n = L&15; j=0..7 -> k = hf*8+j ; j=8..15 -> k = 16+hf*8+(j-8)
__device__ __forceinline__ v16h load_b_frag(const f16* wt, int n, int stride,
                                            int kbase, int hf) {
  const f16* p = wt + n * stride + kbase + hf * 8;
  v8h lo = *(const v8h*)(p);
  v8h hi = *(const v8h*)(p + 16);
  v16h b;
#pragma unroll
  for (int i = 0; i < 8; ++i) { b[i] = lo[i]; b[i + 8] = hi[i]; }
  return b;
}

__device__ __forceinline__ v8h cvt8(float4 a, float4 b) {
  v8h o;
  o[0] = (f16)a.x; o[1] = (f16)a.y; o[2] = (f16)a.z; o[3] = (f16)a.w;
  o[4] = (f16)b.x; o[5] = (f16)b.y; o[6] = (f16)b.z; o[7] = (f16)b.w;
  return o;
}

// 16 x (32*KT) @ aIn  x  WT[64][32*KT]  -> bias+ReLU -> f16 aOut[16][64].
// In-place (aOut may alias aIn): all A-frag DS reads issue before DS stores,
// DS pipeline is in-order per wave.
template <int KT>
__device__ __forceinline__ void gemm_relu_f16(const f16* aIn, int astride,
                                              const f16* wt, int wstride,
                                              const float* bias,
                                              f16* aOut, int lane) {
  const int mm = lane & 15, hf = lane >> 4;
  v16h a[KT];
#pragma unroll
  for (int k = 0; k < KT; ++k) a[k] = load_a_frag(aIn, mm, astride, 32 * k, hf);
#pragma unroll
  for (int t = 0; t < 4; ++t) {
    v8f c;
#pragma unroll
    for (int i = 0; i < 8; ++i) c[i] = 0.f;
#pragma unroll
    for (int k = 0; k < KT; ++k) {
      v16h b = load_b_frag(wt, 16 * t + mm, wstride, 32 * k, hf);
      c = __builtin_amdgcn_wmma_f32_16x16x32_f16(false, a[k], false, b,
                                                 (short)0, c, false, false);
    }
    const float bn = bias[16 * t + mm];
    // C/D layout: lane: n = 16t + (L&15), VGPR r -> row r + 8*(L>>4)
#pragma unroll
    for (int r = 0; r < 8; ++r)
      aOut[(r + 8 * hf) * 64 + 16 * t + mm] = (f16)fmaxf(c[r] + bn, 0.f);
  }
}

// Same GEMM but raw f32 C to LDS (for LayerNorm layers).
template <int KT>
__device__ __forceinline__ void gemm_store(const f16* aIn, int astride,
                                           const f16* wt, int wstride,
                                           float* outLds, int lane) {
  const int mm = lane & 15, hf = lane >> 4;
  v16h a[KT];
#pragma unroll
  for (int k = 0; k < KT; ++k) a[k] = load_a_frag(aIn, mm, astride, 32 * k, hf);
#pragma unroll
  for (int t = 0; t < 4; ++t) {
    v8f c;
#pragma unroll
    for (int i = 0; i < 8; ++i) c[i] = 0.f;
#pragma unroll
    for (int k = 0; k < KT; ++k) {
      v16h b = load_b_frag(wt, 16 * t + mm, wstride, 32 * k, hf);
      c = __builtin_amdgcn_wmma_f32_16x16x32_f16(false, a[k], false, b,
                                                 (short)0, c, false, false);
    }
#pragma unroll
    for (int r = 0; r < 8; ++r)
      outLds[(r + 8 * hf) * 64 + t * 16 + mm] = c[r];
  }
}

// bias + LayerNorm over 64 features, in place on outLds[16][64].
// 1 lane per half-row (float4 vectorized); cross-half reduction via shfl_xor.
__device__ __forceinline__ void epi_bias_ln(float* outLds, const float* bias,
                                            const float* lng, const float* lnb,
                                            int lane) {
  const int r = lane >> 1, hh = lane & 1;
  float4* row = (float4*)(outLds + r * 64 + hh * 32);
  const float4* bp = (const float4*)(bias + hh * 32);
  float4 v[8];
  float s = 0.f, ss = 0.f;
#pragma unroll
  for (int j = 0; j < 8; ++j) {
    float4 a = row[j], b = bp[j];
    a.x += b.x; a.y += b.y; a.z += b.z; a.w += b.w;
    v[j] = a;
    s  += a.x + a.y + a.z + a.w;
    ss += a.x * a.x + a.y * a.y + a.z * a.z + a.w * a.w;
  }
  s  += __shfl_xor(s, 1, 32);
  ss += __shfl_xor(ss, 1, 32);
  float mu  = s * (1.f / 64.f);
  float var = ss * (1.f / 64.f) - mu * mu;
  float inv = rsqrtf(var + LN_EPS);
  const float4* gp  = (const float4*)(lng + hh * 32);
  const float4* bbp = (const float4*)(lnb + hh * 32);
#pragma unroll
  for (int j = 0; j < 8; ++j) {
    float4 a = v[j], g = gp[j], bb = bbp[j];
    a.x = (a.x - mu) * inv * g.x + bb.x;
    a.y = (a.y - mu) * inv * g.y + bb.y;
    a.z = (a.z - mu) * inv * g.z + bb.z;
    a.w = (a.w - mu) * inv * g.w + bb.w;
    row[j] = a;
  }
}

// ---- weight prep: W[K][N] f32 -> WT[Npad][Kpad] f16 (transposed, 0-padded) --
__global__ void prep_wt(const float* __restrict__ W, f16* __restrict__ WT,
                        int K, int Kpad, int N, int Npad) {
  int i = blockIdx.x * 256 + threadIdx.x;
  if (i >= Npad * Kpad) return;
  int n = i / Kpad, k = i % Kpad;
  WT[i] = (k < K && n < N) ? (f16)W[k * N + n] : (f16)0.f;
}

__global__ void zero_f32(float* __restrict__ p, int n) {
  int i = blockIdx.x * 256 + threadIdx.x;
  if (i < n) p[i] = 0.f;
}

// ---- encoder: [R,16] -> MLP(16,64,64,64)+LN -> out [R,64] ------------------
__global__ void __launch_bounds__(32)
enc_kernel(const float* __restrict__ in,
           const f16* w0, const float* b0,
           const f16* w1, const float* b1,
           const f16* w2, const float* b2,
           const float* lng, const float* lnb,
           float* __restrict__ out) {
  __shared__ __align__(16) f16   aS[16 * 64];
  __shared__ __align__(16) float outL[16 * 64];
  const int lane = threadIdx.x;
  const int base = blockIdx.x * 16;
  // stage [16 x 32] f16, cols 16..31 zero (K padded 16->32); 1 lane per row
  if (lane < 16) {
    const float4* p = (const float4*)(in + (size_t)(base + lane) * 16);
    v8h* q = (v8h*)(aS + lane * 32);
    q[0] = cvt8(p[0], p[1]);
    q[1] = cvt8(p[2], p[3]);
    v8h z;
#pragma unroll
    for (int i = 0; i < 8; ++i) z[i] = (f16)0.f;
    q[2] = z; q[3] = z;
  }
  __syncthreads();
  gemm_relu_f16<1>(aS, 32, w0, 32, b0, aS, lane);
  __syncthreads();
  gemm_relu_f16<2>(aS, 64, w1, 64, b1, aS, lane);
  __syncthreads();
  gemm_store<2>(aS, 64, w2, 64, outL, lane);
  __syncthreads();
  epi_bias_ln(outL, b2, lng, lnb, lane);
  __syncthreads();
  {
    const int r = lane >> 1, hh = lane & 1;
    const float4* src = (const float4*)(outL + r * 64 + hh * 32);
    float4* dst = (float4*)(out + (size_t)(base + r) * 64 + hh * 32);
#pragma unroll
    for (int j = 0; j < 8; ++j) dst[j] = src[j];
  }
}

// ---- edge block: concat(h[src],h[dst],e) -> MLP(192,64,64,64)+LN = e_new;
//      agg[dst] += e_new (f32 atomics);  e = e_new + e  ---------------------
__global__ void __launch_bounds__(32)
edge_kernel(const float* __restrict__ h, float* __restrict__ e,
            const int* __restrict__ eidx, int E,
            const f16* w0, const float* b0,
            const f16* w1, const float* b1,
            const f16* w2, const float* b2,
            const float* lng, const float* lnb,
            float* __restrict__ agg) {
  __shared__ __align__(16) f16   aS[16 * 192];
  __shared__ __align__(16) float outL[16 * 64];
  __shared__ __align__(16) float eOld[16 * 64];
  __shared__ int dstSh[16];
  const int lane = threadIdx.x;
  const int mm = lane & 15, hf = lane >> 4;
  const int edge = blockIdx.x * 16 + mm;
  const int s = eidx[edge];
  const int d = eidx[E + edge];
  if (hf == 0) dstSh[mm] = d;
  // gather + f32->f16 stage; each lane: 3 segments x 32 floats (b128 loads)
  const float* segs[3] = { h + (size_t)s * 64, h + (size_t)d * 64,
                           e + (size_t)edge * 64 };
#pragma unroll
  for (int sg = 0; sg < 3; ++sg) {
    const float4* p = (const float4*)(segs[sg] + hf * 32);
    v8h* q = (v8h*)(aS + mm * 192 + sg * 64 + hf * 32);
    float4* eo = (float4*)(eOld + mm * 64 + hf * 32);
#pragma unroll
    for (int j = 0; j < 4; ++j) {
      float4 a = p[2 * j], b = p[2 * j + 1];
      q[j] = cvt8(a, b);
      if (sg == 2) { eo[2 * j] = a; eo[2 * j + 1] = b; }
    }
  }
  __syncthreads();
  gemm_relu_f16<6>(aS, 192, w0, 192, b0, aS, lane);  // 24 WMMA
  __syncthreads();
  gemm_relu_f16<2>(aS, 64, w1, 64, b1, aS, lane);    // 8 WMMA
  __syncthreads();
  gemm_store<2>(aS, 64, w2, 64, outL, lane);         // 8 WMMA
  __syncthreads();
  epi_bias_ln(outL, b2, lng, lnb, lane);
  __syncthreads();
  // scatter-add e_new into agg[dst], residual update e += e_new (vectorized)
  {
    const int r = lane >> 1, hh = lane & 1;
    const int dd = dstSh[r];
    float* ag = agg + (size_t)dd * 64 + hh * 32;
    const float4* vn = (const float4*)(outL + r * 64 + hh * 32);
    const float4* eo = (const float4*)(eOld + r * 64 + hh * 32);
    float4* eout = (float4*)(e + (size_t)(blockIdx.x * 16 + r) * 64 + hh * 32);
#pragma unroll
    for (int j = 0; j < 8; ++j) {
      float4 v = vn[j], o = eo[j];
      atomicAdd(ag + 4 * j + 0, v.x);
      atomicAdd(ag + 4 * j + 1, v.y);
      atomicAdd(ag + 4 * j + 2, v.z);
      atomicAdd(ag + 4 * j + 3, v.w);
      float4 rr;
      rr.x = v.x + o.x; rr.y = v.y + o.y; rr.z = v.z + o.z; rr.w = v.w + o.w;
      eout[j] = rr;
    }
  }
}

// ---- node block: concat(h,agg) -> MLP(128,64,64,64)+LN; h += out ----------
__global__ void __launch_bounds__(32)
node_kernel(float* __restrict__ hBuf, const float* __restrict__ agg,
            const f16* w0, const float* b0,
            const f16* w1, const float* b1,
            const f16* w2, const float* b2,
            const float* lng, const float* lnb) {
  __shared__ __align__(16) f16   aS[16 * 128];
  __shared__ __align__(16) float outL[16 * 64];
  __shared__ __align__(16) float hOld[16 * 64];
  const int lane = threadIdx.x;
  const int mm = lane & 15, hf = lane >> 4;
  const size_t node = (size_t)blockIdx.x * 16 + mm;
  {
    const float4* ph = (const float4*)(hBuf + node * 64 + hf * 32);
    const float4* pa = (const float4*)(agg  + node * 64 + hf * 32);
    v8h* q0 = (v8h*)(aS + mm * 128 + hf * 32);
    v8h* q1 = (v8h*)(aS + mm * 128 + 64 + hf * 32);
    float4* ho = (float4*)(hOld + mm * 64 + hf * 32);
#pragma unroll
    for (int j = 0; j < 4; ++j) {
      float4 a = ph[2 * j], b = ph[2 * j + 1];
      q0[j] = cvt8(a, b);
      ho[2 * j] = a; ho[2 * j + 1] = b;
      q1[j] = cvt8(pa[2 * j], pa[2 * j + 1]);
    }
  }
  __syncthreads();
  gemm_relu_f16<4>(aS, 128, w0, 128, b0, aS, lane);  // 16 WMMA
  __syncthreads();
  gemm_relu_f16<2>(aS, 64, w1, 64, b1, aS, lane);
  __syncthreads();
  gemm_store<2>(aS, 64, w2, 64, outL, lane);
  __syncthreads();
  epi_bias_ln(outL, b2, lng, lnb, lane);
  __syncthreads();
  {
    const int base = blockIdx.x * 16;
    const int r = lane >> 1, hh = lane & 1;
    const float4* vn = (const float4*)(outL + r * 64 + hh * 32);
    const float4* ho = (const float4*)(hOld + r * 64 + hh * 32);
    float4* dst = (float4*)(hBuf + (size_t)(base + r) * 64 + hh * 32);
#pragma unroll
    for (int j = 0; j < 8; ++j) {
      float4 v = vn[j], o = ho[j];
      float4 rr;
      rr.x = v.x + o.x; rr.y = v.y + o.y; rr.z = v.z + o.z; rr.w = v.w + o.w;
      dst[j] = rr;
    }
  }
}

// ---- decoder: [R,64] -> MLP(64,64,64,8) -> out [R,8] ----------------------
__global__ void __launch_bounds__(32)
dec_kernel(const float* __restrict__ in,
           const f16* w0, const float* b0,
           const f16* w1, const float* b1,
           const f16* w2pad /*[16][64], cols 8..15 zero*/, const float* b2 /*[8]*/,
           float* __restrict__ out) {
  __shared__ __align__(16) f16 aS[16 * 64];
  const int lane = threadIdx.x;
  const int mm = lane & 15, hf = lane >> 4;
  const int base = blockIdx.x * 16;
  {
    const float4* p = (const float4*)(in + (size_t)base * 64) + lane * 8;
    v8h* q = (v8h*)(aS + lane * 32);
#pragma unroll
    for (int j = 0; j < 4; ++j) q[j] = cvt8(p[2 * j], p[2 * j + 1]);
  }
  __syncthreads();
  gemm_relu_f16<2>(aS, 64, w0, 64, b0, aS, lane);
  __syncthreads();
  gemm_relu_f16<2>(aS, 64, w1, 64, b1, aS, lane);
  __syncthreads();
  // final 64->8 as a single N-tile WMMA (cols 8..15 of B zero-padded)
  v16h a0 = load_a_frag(aS, mm, 64, 0, hf);
  v16h a1 = load_a_frag(aS, mm, 64, 32, hf);
  v16h bb0 = load_b_frag(w2pad, mm, 64, 0, hf);
  v16h bb1 = load_b_frag(w2pad, mm, 64, 32, hf);
  v8f c;
#pragma unroll
  for (int i = 0; i < 8; ++i) c[i] = 0.f;
  c = __builtin_amdgcn_wmma_f32_16x16x32_f16(false, a0, false, bb0, (short)0, c, false, false);
  c = __builtin_amdgcn_wmma_f32_16x16x32_f16(false, a1, false, bb1, (short)0, c, false, false);
  if (mm < 8) {
#pragma unroll
    for (int r = 0; r < 8; ++r)
      out[(size_t)(base + r + 8 * hf) * 8 + mm] = c[r] + b2[mm];
  }
}

// ============================================================================
extern "C" void kernel_launch(void* const* d_in, const int* in_sizes, int n_in,
                              void* d_out, int out_size, void* d_ws, size_t ws_size,
                              hipStream_t stream) {
  (void)in_sizes; (void)n_in; (void)out_size; (void)ws_size;
  const int NN = 50000, NE = 800000;
  const float* x     = (const float*)d_in[0];
  const int*   eidx  = (const int*)d_in[1];
  const float* efeat = (const float*)d_in[2];
  auto F = [&](int i) -> const float* { return (const float*)d_in[i]; };

  // ---- weight spec table (30 matrices) ----
  int widx[30], wK[30], wKp[30], wN[30], wNp[30];
  int c = 0;
  for (int b = 0; b < 3; ++b) {
    int eb = 3 + 16 * b, nb = eb + 8;
    widx[c]=eb+0; wK[c]=192; wKp[c]=192; wN[c]=64; wNp[c]=64; ++c;
    widx[c]=eb+1; wK[c]=64;  wKp[c]=64;  wN[c]=64; wNp[c]=64; ++c;
    widx[c]=eb+2; wK[c]=64;  wKp[c]=64;  wN[c]=64; wNp[c]=64; ++c;
    widx[c]=nb+0; wK[c]=128; wKp[c]=128; wN[c]=64; wNp[c]=64; ++c;
    widx[c]=nb+1; wK[c]=64;  wKp[c]=64;  wN[c]=64; wNp[c]=64; ++c;
    widx[c]=nb+2; wK[c]=64;  wKp[c]=64;  wN[c]=64; wNp[c]=64; ++c;
  }
  const int decb[2] = {51, 57};              // dec_edge, dec_node
  for (int t = 0; t < 2; ++t) {
    int b0 = decb[t];
    widx[c]=b0+0; wK[c]=64; wKp[c]=64; wN[c]=64; wNp[c]=64; ++c;
    widx[c]=b0+1; wK[c]=64; wKp[c]=64; wN[c]=64; wNp[c]=64; ++c;
    widx[c]=b0+2; wK[c]=64; wKp[c]=64; wN[c]=8;  wNp[c]=16; ++c;
  }
  const int encb[2] = {63, 71};              // enc_edge, enc_node
  for (int t = 0; t < 2; ++t) {
    int b0 = encb[t];
    widx[c]=b0+0; wK[c]=16; wKp[c]=32; wN[c]=64; wNp[c]=64; ++c;
    widx[c]=b0+1; wK[c]=64; wKp[c]=64; wN[c]=64; wNp[c]=64; ++c;
    widx[c]=b0+2; wK[c]=64; wKp[c]=64; wN[c]=64; wNp[c]=64; ++c;
  }

  // ---- workspace carve-up ----
  char* ws = (char*)d_ws;
  size_t off = 0;
  f16* wptr[30];
  for (int i = 0; i < 30; ++i) {
    size_t elems = (size_t)wNp[i] * wKp[i];
    wptr[i] = (f16*)(ws + off);
    off += ((elems * sizeof(f16) + 255) & ~(size_t)255);
    int total = (int)elems;
    prep_wt<<<(total + 255) / 256, 256, 0, stream>>>(F(widx[i]), wptr[i],
                                                     wK[i], wKp[i], wN[i], wNp[i]);
  }
  off = (off + 255) & ~(size_t)255;
  float* hB   = (float*)(ws + off); off += (size_t)NN * 64 * sizeof(float);
  float* eB   = (float*)(ws + off); off += (size_t)NE * 64 * sizeof(float);
  float* aggB = (float*)(ws + off); off += (size_t)NN * 64 * sizeof(float);

  // wptr layout: blocks 6b+{eW0,eW1,eW2,nW0,nW1,nW2}; 18..20 dec_edge;
  // 21..23 dec_node; 24..26 enc_edge; 27..29 enc_node
  enc_kernel<<<NN / 16, 32, 0, stream>>>(x,
      wptr[27], F(74), wptr[28], F(75), wptr[29], F(76), F(78), F(77), hB);
  enc_kernel<<<NE / 16, 32, 0, stream>>>(efeat,
      wptr[24], F(66), wptr[25], F(67), wptr[26], F(68), F(70), F(69), eB);

  for (int it = 0; it < 6; ++it) {           // REPEAT(2) x N_BLOCKS(3)
    int b = it % 3, eb = 3 + 16 * b, nb = eb + 8;
    zero_f32<<<(NN * 64 + 255) / 256, 256, 0, stream>>>(aggB, NN * 64);
    edge_kernel<<<NE / 16, 32, 0, stream>>>(hB, eB, eidx, NE,
        wptr[6 * b + 0], F(eb + 3), wptr[6 * b + 1], F(eb + 4),
        wptr[6 * b + 2], F(eb + 5), F(eb + 7), F(eb + 6), aggB);
    node_kernel<<<NN / 16, 32, 0, stream>>>(hB, aggB,
        wptr[6 * b + 3], F(nb + 3), wptr[6 * b + 4], F(nb + 4),
        wptr[6 * b + 5], F(nb + 5), F(nb + 7), F(nb + 6));
  }

  float* outp = (float*)d_out;               // [node_out | edge_out]
  dec_kernel<<<NN / 16, 32, 0, stream>>>(hB,
      wptr[21], F(60), wptr[22], F(61), wptr[23], F(62), outp);
  dec_kernel<<<NE / 16, 32, 0, stream>>>(eB,
      wptr[18], F(54), wptr[19], F(55), wptr[20], F(56), outp + (size_t)NN * 8);
}
